// GATv2MiniModel_42004780155066
// MI455X (gfx1250) — matrix operand order repaired and checked by reading, not compiled, not run on hardware
//
#include <hip/hip_runtime.h>
#include <math.h>

typedef __bf16 bf16;
typedef bf16  v16bf __attribute__((ext_vector_type(16)));
typedef float v8f   __attribute__((ext_vector_type(8)));

#define NEG_SLOPE 0.2f

// ---------------------------------------------------------------------------
// Weight pre-pass: Wt[n*K + k] = (bf16) W[k*N + n]   (transpose + f32->bf16)
// ---------------------------------------------------------------------------
__global__ __launch_bounds__(256)
void k_transpose_bf16(const float* __restrict__ W, bf16* __restrict__ Wt, int K, int N) {
    int i = blockIdx.x * blockDim.x + threadIdx.x;
    if (i >= K * N) return;
    int n = i / K, k = i - n * K;
    Wt[(size_t)n * K + k] = (bf16)W[(size_t)k * N + n];
}

// ---------------------------------------------------------------------------
// WMMA GEMM: C[M,N] = act(A[M,K] @ W + bias[N]), W given transposed bf16 [N,K].
// One block owns a full 16-row x N stripe: blockDim = (N/64)*32, wave w covers
// columns [w*64, w*64+64). The 16x32 A tile is staged once per k-step into LDS,
// already converted to bf16 and swizzled into per-lane A-fragment order
// (lane L: m=L%16, kbase=(L<16)?0:16; elems 0..15 = A[m][kb..kb+15]),
// so every lane issues one aligned 32B
// ds_load and A is read from HBM exactly once. Double-buffered: staging of
// k-step i+1 overlaps the 4 WMMAs of step i; one barrier per step.
// B-fragment (16-bit 32x16): lane L -> col N=L%16, kbase=(L<16)?0:16,
//   elems 0..15 = Wt[n][kb..kb+15] (contiguous; L2-resident weights).
// C layout: acc elem r -> row row0 + r + ((L<16)?0:8), col = L%16.
// ---------------------------------------------------------------------------
__global__ __launch_bounds__(256)
void k_gemm_wmma(const float* __restrict__ A, const bf16* __restrict__ Wt,
                 const float* __restrict__ bias, float* __restrict__ C,
                 int M, int K, int N, int relu)
{
    __shared__ __attribute__((aligned(64))) bf16 smem[2][512];
    const int lane = threadIdx.x & 31;
    const int wv   = threadIdx.x >> 5;
    const int row0 = blockIdx.x * 16;
    const int col0 = wv * 64;
    const int nlo  = lane & 15;
    const int kB   = (lane < 16) ? 0 : 16;

    // stage k-step 0 (bf16, fragment-order swizzle)
    for (int s = threadIdx.x; s < 512; s += blockDim.x) {
        const int L = s >> 4, j = s & 15;
        const int m = L & 15;
        const int k = ((L & 16) ? 16 : 0) + j;
        smem[0][s] = (bf16)A[(size_t)(row0 + m) * K + k];
    }
    __syncthreads();

    v8f acc[4] = {};
    const int nsteps = K >> 5;
    for (int i = 0; i < nsteps; ++i) {
        const v16bf af = *(const v16bf*)&smem[i & 1][lane * 16];
        if (i + 1 < nsteps) {            // overlap staging of next k-step
            const int kk = (i + 1) << 5;
            for (int s = threadIdx.x; s < 512; s += blockDim.x) {
                const int L = s >> 4, j = s & 15;
                const int m = L & 15;
                const int k = ((L & 16) ? 16 : 0) + j;
                smem[(i + 1) & 1][s] = (bf16)A[(size_t)(row0 + m) * K + kk + k];
            }
        }
        const int kk = i << 5;
        #pragma unroll
        for (int t = 0; t < 4; ++t) {
            const int n = col0 + t * 16 + nlo;
            const v16bf bfrag = *(const v16bf*)(Wt + (size_t)n * K + kk + kB);
            acc[t] = __builtin_amdgcn_wmma_f32_16x16x32_bf16(
                false, af, false, bfrag, (short)0, acc[t], false, false);
        }
        __syncthreads();
    }

    const int rbase = row0 + ((lane < 16) ? 0 : 8);
    #pragma unroll
    for (int t = 0; t < 4; ++t) {
        const int n = col0 + t * 16 + nlo;
        const float bv = bias ? bias[n] : 0.0f;
        #pragma unroll
        for (int r = 0; r < 8; ++r) {
            float v = acc[t][r] + bv;
            if (relu) v = fmaxf(v, 0.0f);
            C[(size_t)(rbase + r) * N + n] = v;
        }
    }
}

// ---------------------------------------------------------------------------
// Utility / edge-phase kernels
// ---------------------------------------------------------------------------
__global__ __launch_bounds__(256)
void k_fill(float* __restrict__ p, float v, size_t n) {
    size_t i  = (size_t)blockIdx.x * blockDim.x + threadIdx.x;
    size_t st = (size_t)gridDim.x * blockDim.x;
    for (; i < n; i += st) p[i] = v;
}

__device__ inline void atomicMaxF(float* addr, float v) {
    if (v >= 0.0f) atomicMax((int*)addr, __float_as_int(v));
    else           atomicMin((unsigned int*)addr, (unsigned int)__float_as_int(v));
}

// Wave per (edge, head): logits[e,h] = sum_c leaky(xl[src]+xr[dst]+er*We) * att
__global__ __launch_bounds__(256)
void k_logits(const float* __restrict__ xl, const float* __restrict__ xr,
              const float* __restrict__ ea, const float* __restrict__ logw,
              const float* __restrict__ We, const float* __restrict__ att,
              const int* __restrict__ src, const int* __restrict__ dst,
              float* __restrict__ logits, float* __restrict__ mbuf, int E)
{
    const int wid  = (blockIdx.x * blockDim.x + threadIdx.x) >> 5;
    const int lane = threadIdx.x & 31;
    if (wid >= E * 4) return;
    const int e = wid >> 2, h = wid & 3;
    const int s = src[e], d = dst[e];
    const float er = ea[e] * __expf(logw[0]);
    const int co = h * 128 + lane * 4;
    const float4 xlv = *(const float4*)(xl + (size_t)s * 512 + co);
    const float4 xrv = *(const float4*)(xr + (size_t)d * 512 + co);
    const float4 wev = *(const float4*)(We + co);
    const float4 atv = *(const float4*)(att + co);
    float g, dot = 0.0f;
    g = xlv.x + xrv.x + er * wev.x; g = g > 0.f ? g : NEG_SLOPE * g; dot += g * atv.x;
    g = xlv.y + xrv.y + er * wev.y; g = g > 0.f ? g : NEG_SLOPE * g; dot += g * atv.y;
    g = xlv.z + xrv.z + er * wev.z; g = g > 0.f ? g : NEG_SLOPE * g; dot += g * atv.z;
    g = xlv.w + xrv.w + er * wev.w; g = g > 0.f ? g : NEG_SLOPE * g; dot += g * atv.w;
    for (int o = 16; o; o >>= 1) dot += __shfl_xor(dot, o, 32);
    if (lane == 0) {
        logits[wid] = dot;
        atomicMaxF(&mbuf[(size_t)d * 4 + h], dot);
    }
}

// Thread per (edge, head): p = exp(logit - m[dst]); den[dst] += p
__global__ __launch_bounds__(256)
void k_pexp(const float* __restrict__ logits, const int* __restrict__ dst,
            const float* __restrict__ mbuf, float* __restrict__ p,
            float* __restrict__ den, int E)
{
    int i = blockIdx.x * blockDim.x + threadIdx.x;
    if (i >= E * 4) return;
    int e = i >> 2, h = i & 3;
    int d = dst[e];
    float pv = __expf(logits[i] - mbuf[(size_t)d * 4 + h]);
    p[i] = pv;
    atomicAdd(&den[(size_t)d * 4 + h], pv);
}

// Wave per (edge, head): agg[dst,h,:] += coef * xl[src,h,:]
__global__ __launch_bounds__(256)
void k_agg(const float* __restrict__ xl, const float* __restrict__ p,
           const float* __restrict__ den, const int* __restrict__ src,
           const int* __restrict__ dst, float* __restrict__ agg, int E)
{
    const int wid  = (blockIdx.x * blockDim.x + threadIdx.x) >> 5;
    const int lane = threadIdx.x & 31;
    if (wid >= E * 4) return;
    const int e = wid >> 2, h = wid & 3;
    const int s = src[e], d = dst[e];
    const float coef = p[wid] / fmaxf(den[(size_t)d * 4 + h], 1e-16f);
    const int co = h * 128 + lane * 4;
    const float4 xlv = *(const float4*)(xl + (size_t)s * 512 + co);
    float* ag = agg + (size_t)d * 512 + co;
    atomicAdd(ag + 0, coef * xlv.x);
    atomicAdd(ag + 1, coef * xlv.y);
    atomicAdd(ag + 2, coef * xlv.z);
    atomicAdd(ag + 3, coef * xlv.w);
}

// Thread per (node, c): mean over 4 heads + bias (+ optional relu)
__global__ __launch_bounds__(256)
void k_head_mean(const float* __restrict__ agg, const float* __restrict__ bias,
                 float* __restrict__ out, int Nd, int relu)
{
    int i = blockIdx.x * blockDim.x + threadIdx.x;
    if (i >= Nd * 128) return;
    int dnode = i >> 7, c = i & 127;
    const float* a = agg + (size_t)dnode * 512 + c;
    float v = 0.25f * (a[0] + a[128] + a[256] + a[384]) + bias[c];
    out[i] = relu ? fmaxf(v, 0.0f) : v;
}

// Wave per row: L2 normalize 128-wide rows
__global__ __launch_bounds__(256)
void k_l2norm(const float* __restrict__ in, float* __restrict__ out, int rows) {
    const int wid  = (blockIdx.x * blockDim.x + threadIdx.x) >> 5;
    const int lane = threadIdx.x & 31;
    if (wid >= rows) return;
    const float4 v = *(const float4*)(in + (size_t)wid * 128 + lane * 4);
    float s = v.x * v.x + v.y * v.y + v.z * v.z + v.w * v.w;
    for (int o = 16; o; o >>= 1) s += __shfl_xor(s, o, 32);
    const float inv = 1.0f / fmaxf(sqrtf(s), 1e-12f);
    float4 r; r.x = v.x * inv; r.y = v.y * inv; r.z = v.z * inv; r.w = v.w * inv;
    *(float4*)(out + (size_t)wid * 128 + lane * 4) = r;
}

// ---------------------------------------------------------------------------
// Host orchestration
// ---------------------------------------------------------------------------
extern "C" void kernel_launch(void* const* d_in, const int* in_sizes, int n_in,
                              void* d_out, int out_size, void* d_ws, size_t ws_size,
                              hipStream_t stream)
{
    const float* x_user    = (const float*)d_in[0];
    const float* x_item    = (const float*)d_in[1];
    const int*   src_rates = (const int*)d_in[2];
    const int*   dst_rates = (const int*)d_in[3];
    const int*   src_rated = (const int*)d_in[4];
    const int*   dst_rated = (const int*)d_in[5];
    const float* ea_rates  = (const float*)d_in[6];
    const float* ea_rated  = (const float*)d_in[7];
    const float* Wp_user   = (const float*)d_in[8];
    const float* bp_user   = (const float*)d_in[9];
    const float* Wp_item   = (const float*)d_in[10];
    const float* bp_item   = (const float*)d_in[11];
    const float* Wl_rates  = (const float*)d_in[12];
    const float* bl_rates  = (const float*)d_in[13];
    const float* Wr_rates  = (const float*)d_in[14];
    const float* br_rates  = (const float*)d_in[15];
    const float* We_rates  = (const float*)d_in[16];
    const float* att_rates = (const float*)d_in[17];
    const float* bias_rates= (const float*)d_in[18];
    const float* logw_rates= (const float*)d_in[19];
    const float* Wl_rated  = (const float*)d_in[20];
    const float* bl_rated  = (const float*)d_in[21];
    const float* Wr_rated  = (const float*)d_in[22];
    const float* br_rated  = (const float*)d_in[23];
    const float* We_rated  = (const float*)d_in[24];
    const float* att_rated = (const float*)d_in[25];
    const float* bias_rated= (const float*)d_in[26];
    const float* logw_rated= (const float*)d_in[27];
    const float* Wf_user   = (const float*)d_in[28];
    const float* bf_user   = (const float*)d_in[29];
    const float* Wf_item   = (const float*)d_in[30];
    const float* bf_item   = (const float*)d_in[31];

    const int Nu = in_sizes[0] / 128;   // 100000
    const int Ni = in_sizes[1] / 64;    // 50000
    const int E  = in_sizes[2];         // 200000

    // workspace carve-out (256B aligned)
    size_t off = 0;
    auto alloc = [&](size_t bytes) -> void* {
        void* p = (char*)d_ws + off;
        off += (bytes + 255) & ~(size_t)255;
        return p;
    };
    float* hu_buf[2]; float* hi_buf[2];
    hu_buf[0] = (float*)alloc((size_t)Nu * 128 * 4);
    hu_buf[1] = (float*)alloc((size_t)Nu * 128 * 4);
    hi_buf[0] = (float*)alloc((size_t)Ni * 128 * 4);
    hi_buf[1] = (float*)alloc((size_t)Ni * 128 * 4);
    float* xlb    = (float*)alloc((size_t)Nu * 512 * 4);
    float* xrb    = (float*)alloc((size_t)Nu * 512 * 4);   // reused as agg after logits
    float* logits = (float*)alloc((size_t)E * 4 * 4);
    float* pbuf   = (float*)alloc((size_t)E * 4 * 4);
    float* mbuf   = (float*)alloc((size_t)Nu * 4 * 4);
    float* den    = (float*)alloc((size_t)Nu * 4 * 4);
    bf16* WpT_user = (bf16*)alloc((size_t)128 * 128 * 2);
    bf16* WpT_item = (bf16*)alloc((size_t)128 * 64 * 2);
    bf16* WlT_rates = (bf16*)alloc((size_t)2 * 512 * 128 * 2);
    bf16* WrT_rates = (bf16*)alloc((size_t)2 * 512 * 128 * 2);
    bf16* WlT_rated = (bf16*)alloc((size_t)2 * 512 * 128 * 2);
    bf16* WrT_rated = (bf16*)alloc((size_t)2 * 512 * 128 * 2);
    bf16* WfT_user = (bf16*)alloc((size_t)128 * 128 * 2);
    bf16* WfT_item = (bf16*)alloc((size_t)128 * 128 * 2);

    auto tr = [&](const float* W, bf16* Wt, int K, int N) {
        k_transpose_bf16<<<(K * N + 255) / 256, 256, 0, stream>>>(W, Wt, K, N);
    };
    auto gemm = [&](const float* A, const bf16* Wt, const float* b, float* Co,
                    int M, int K, int N, int relu) {
        dim3 grid(M / 16);
        dim3 block((N / 64) * 32);     // one wave per 64-col slice, full N stripe
        k_gemm_wmma<<<grid, block, 0, stream>>>(A, Wt, b, Co, M, K, N, relu);
    };
    auto fill = [&](float* p, float v, size_t n) {
        int blocks = (int)((n + 255) / 256);
        k_fill<<<blocks, 256, 0, stream>>>(p, v, n);
    };

    // weight pre-pass
    tr(Wp_user, WpT_user, 128, 128);
    tr(Wp_item, WpT_item, 64, 128);
    for (int l = 0; l < 2; ++l) {
        tr(Wl_rates + (size_t)l * 128 * 512, WlT_rates + (size_t)l * 512 * 128, 128, 512);
        tr(Wr_rates + (size_t)l * 128 * 512, WrT_rates + (size_t)l * 512 * 128, 128, 512);
        tr(Wl_rated + (size_t)l * 128 * 512, WlT_rated + (size_t)l * 512 * 128, 128, 512);
        tr(Wr_rated + (size_t)l * 128 * 512, WrT_rated + (size_t)l * 512 * 128, 128, 512);
    }
    tr(Wf_user, WfT_user, 128, 128);
    tr(Wf_item, WfT_item, 128, 128);

    // input projections (relu)
    gemm(x_user, WpT_user, bp_user, hu_buf[0], Nu, 128, 128, 1);
    gemm(x_item, WpT_item, bp_item, hi_buf[0], Ni, 64, 128, 1);

    const int edgeWaveBlocks = (E * 4 * 32 + 255) / 256;
    auto gat = [&](const float* hsrc, int Nsrc, const float* hdst, int Ndst,
                   const int* src, const int* dst, const float* ea, const float* logw,
                   const bf16* WlT, const float* bl, const bf16* WrT, const float* br,
                   const float* We, const float* att, const float* bias, float* outbuf) {
        gemm(hsrc, WlT, bl, xlb, Nsrc, 128, 512, 0);            // xl [Nsrc,512]
        gemm(hdst, WrT, br, xrb, Ndst, 128, 512, 0);            // xr [Ndst,512]
        fill(mbuf, -INFINITY, (size_t)Ndst * 4);
        fill(den, 0.0f, (size_t)Ndst * 4);
        k_logits<<<edgeWaveBlocks, 256, 0, stream>>>(xlb, xrb, ea, logw, We, att,
                                                     src, dst, logits, mbuf, E);
        fill(xrb, 0.0f, (size_t)Ndst * 512);                    // xr dead -> agg
        k_pexp<<<(E * 4 + 255) / 256, 256, 0, stream>>>(logits, dst, mbuf, pbuf, den, E);
        k_agg<<<edgeWaveBlocks, 256, 0, stream>>>(xlb, pbuf, den, src, dst, xrb, E);
        k_head_mean<<<(Ndst * 128 + 255) / 256, 256, 0, stream>>>(xrb, bias, outbuf, Ndst, 1);
    };

    int cur = 0;
    for (int l = 0; l < 2; ++l) {
        const int nxt = cur ^ 1;
        gat(hu_buf[cur], Nu, hi_buf[cur], Ni, src_rates, dst_rates, ea_rates, logw_rates,
            WlT_rates + (size_t)l * 512 * 128, bl_rates + (size_t)l * 512,
            WrT_rates + (size_t)l * 512 * 128, br_rates + (size_t)l * 512,
            We_rates + (size_t)l * 512, att_rates + (size_t)l * 512,
            bias_rates + (size_t)l * 128, hi_buf[nxt]);
        gat(hi_buf[cur], Ni, hu_buf[cur], Nu, src_rated, dst_rated, ea_rated, logw_rated,
            WlT_rated + (size_t)l * 512 * 128, bl_rated + (size_t)l * 512,
            WrT_rated + (size_t)l * 512 * 128, br_rated + (size_t)l * 512,
            We_rated + (size_t)l * 512, att_rated + (size_t)l * 512,
            bias_rated + (size_t)l * 128, hu_buf[nxt]);
        cur = nxt;
    }

    // final projections + l2norm into d_out (zu then zi)
    float* out_f = (float*)d_out;
    gemm(hu_buf[cur], WfT_user, bf_user, xlb, Nu, 128, 128, 0);
    k_l2norm<<<(Nu * 32 + 255) / 256, 256, 0, stream>>>(xlb, out_f, Nu);
    gemm(hi_buf[cur], WfT_item, bf_item, xlb, Ni, 128, 128, 0);
    k_l2norm<<<(Ni * 32 + 255) / 256, 256, 0, stream>>>(xlb, out_f + (size_t)Nu * 128, Ni);
}
